// SimMIM_patch2_seg_chn_37993280700716
// MI455X (gfx1250) — matrix operand order repaired and checked by reading, not compiled, not run on hardware
//
#include <hip/hip_runtime.h>
#include <cstdint>

typedef __attribute__((ext_vector_type(16))) _Float16 v16h;
typedef __attribute__((ext_vector_type(8)))  _Float16 v8h;
typedef __attribute__((ext_vector_type(8)))  float    v8f;
typedef __attribute__((ext_vector_type(4)))  int      v4i;

#define B_    8
#define NTOK  1800
#define D_    512
#define NH_   8
#define DH_   64
#define MLP_  2048
#define NM_   900
#define PD_   6
#define L_    4
#define BN_   (B_*NTOK)      // 14400
#define LOSS_SCALE (1.0f/38880000.0f)  // 1/(B*NM*PD*NM)

// ---- gfx1250 async global->LDS (builtins confirmed present by round-2 probe) ----
#if defined(__HIP_DEVICE_COMPILE__) && __has_builtin(__builtin_amdgcn_global_load_async_to_lds_b128) && __has_builtin(__builtin_amdgcn_s_wait_asynccnt)
#define HAVE_ASYNC_LDS 1
#else
#define HAVE_ASYNC_LDS 0
#endif

#if HAVE_ASYNC_LDS
typedef __attribute__((address_space(1))) v4i as1_v4i;   // global
typedef __attribute__((address_space(3))) v4i as3_v4i;   // LDS
// copy 16 bytes global -> LDS via the async DMA path (tracked by ASYNCcnt)
__device__ __forceinline__ void async_copy16(const _Float16* gsrc, _Float16* ldst) {
  __builtin_amdgcn_global_load_async_to_lds_b128(
      (as1_v4i*)(uintptr_t)gsrc,
      (as3_v4i*)(uint32_t)(uintptr_t)ldst,   // low 32 bits of generic shared ptr = LDS offset
      0, 0);
}
#endif

// ---------------- WMMA helpers (CDNA5 16x16x32 f16 -> f32) ----------------

__device__ __forceinline__ v8f wmma_f16(v16h a, v16h b, v8f c) {
  return __builtin_amdgcn_wmma_f32_16x16x32_f16(false, a, false, b, (short)0, c, false, false);
}

__device__ __forceinline__ v16h make_v16(v8h lo, v8h hi) {
  return __builtin_shufflevector(lo, hi, 0,1,2,3,4,5,6,7,8,9,10,11,12,13,14,15);
}

// A fragment: 16x32 (MxK), row-major source [rows, ld] f16.
// lane 0-15: rows M=0..15, halves 0..7 = K=k0..k0+7, halves 8..15 = K=k0+16..k0+23
// lane 16-31: same rows, koff +8.
__device__ __forceinline__ v16h load_a_frag(const _Float16* base, int ld, int row0, int k0) {
  int lane = threadIdx.x & 31;
  int r = row0 + (lane & 15);
  int koff = (lane & 16) ? 8 : 0;
  const _Float16* p = base + (size_t)r * ld + k0 + koff;
  return make_v16(*(const v8h*)p, *(const v8h*)(p + 16));
}
__device__ __forceinline__ v16h load_a_frag_clamp(const _Float16* base, int ld, int row0, int k0, int rmax) {
  int lane = threadIdx.x & 31;
  int r = row0 + (lane & 15); if (r > rmax) r = rmax;
  int koff = (lane & 16) ? 8 : 0;
  const _Float16* p = base + (size_t)r * ld + k0 + koff;
  return make_v16(*(const v8h*)p, *(const v8h*)(p + 16));
}
// B fragment: 32x16 (KxN). Source is W^T stored [N, K] row-major, so column n of B
// is the contiguous row n. lane 0-15: col N=0..15, halves 0..7 = K=k0..k0+7, halves 8..15 = K=k0+16..k0+23;
// lanes 16-31: koff +8.
__device__ __forceinline__ v16h load_b_frag(const _Float16* base, int ld, int n0, int k0) {
  int lane = threadIdx.x & 31;
  int r = n0 + (lane & 15);
  int koff = (lane & 16) ? 8 : 0;
  const _Float16* p = base + (size_t)r * ld + k0 + koff;
  return make_v16(*(const v8h*)p, *(const v8h*)(p + 16));
}

__device__ __forceinline__ float gelu_f(float x) {
  float x3 = x * x * x;
  return 0.5f * x * (1.0f + tanhf(0.7978845608028654f * (x + 0.044715f * x3)));
}

// ---------------- setup kernels ----------------

__global__ void init_kernel(int* mask, float* out, int n) {
  int t = blockIdx.x * blockDim.x + threadIdx.x;
  if (t < n) mask[t] = 0;
  if (t == 0) out[0] = 0.0f;
}

__global__ void scatter_mask_kernel(int* mask, const int* mIdx) {
  int t = blockIdx.x * blockDim.x + threadIdx.x;
  if (t < B_ * NM_) {
    int b = t / NM_;
    mask[b * NTOK + mIdx[t]] = 1;
  }
}

// in: [L, K, Nout] f32  ->  out: [L, Nout, K] f16 (transposed per layer)
__global__ void wtrans_kernel(const float* in, _Float16* out, int K, int Nout, int total) {
  int kn = K * Nout;
  for (int idx = blockIdx.x * blockDim.x + threadIdx.x; idx < total;
       idx += gridDim.x * blockDim.x) {
    int l = idx / kn;
    int r = idx - l * kn;
    int k = r / Nout;
    int n = r - k * Nout;
    out[(size_t)l * kn + (size_t)n * K + k] = (_Float16)in[idx];
  }
}

// ---------------- embedding ----------------
__global__ void __launch_bounds__(128)
embed_kernel(const float* img, const float* pos_table, const float* val_table,
             const float* patch_W, const float* patch_b, const float* mask_token,
             const int* vlen, const int* mask, float* x, float* patches) {
  int bn = blockIdx.x;               // b*NTOK + n
  int b = bn / NTOK, n = bn - b * NTOK;
  int hj = n / 75, ws = n - hj * 75;
  float p[PD_];
#pragma unroll
  for (int t = 0; t < 2; ++t)
#pragma unroll
    for (int c = 0; c < 3; ++c)
      p[t * 3 + c] = img[(((size_t)b * 3 + c) * 24 + hj) * 150 + ws * 2 + t];
  int vl = vlen[b]; if (vl == 150) vl = 149;
  int nv = (vl + 1) >> 1;
  int seg = (ws < nv) ? 1 : 0;
  int msk = mask[bn];
  int tid = threadIdx.x;
#pragma unroll
  for (int i = 0; i < 4; ++i) {
    int d = i * 128 + tid;
    float pos = pos_table[(size_t)(n + 1) * D_ + d] + val_table[(size_t)seg * D_ + d];
    float val;
    if (msk) {
      val = mask_token[d] + pos;
    } else {
      float t = patch_b[d];
#pragma unroll
      for (int pp = 0; pp < PD_; ++pp) t += p[pp] * patch_W[pp * D_ + d];
      val = t + pos;
    }
    x[(size_t)bn * D_ + d] = val;
  }
  if (tid < PD_) patches[(size_t)bn * PD_ + tid] = p[tid];
}

// ---------------- layernorm (f32 in -> f16 out) ----------------
__global__ void __launch_bounds__(128)
ln_kernel(const float* x, const float* gs, const float* gb, _Float16* out) {
  __shared__ float red[128];
  int row = blockIdx.x, tid = threadIdx.x;
  const float* xr = x + (size_t)row * D_;
  float v[4];
  float s = 0.0f;
#pragma unroll
  for (int i = 0; i < 4; ++i) { v[i] = xr[i * 128 + tid]; s += v[i]; }
  red[tid] = s; __syncthreads();
  for (int off = 64; off > 0; off >>= 1) { if (tid < off) red[tid] += red[tid + off]; __syncthreads(); }
  float mean = red[0] * (1.0f / D_);
  __syncthreads();
  float s2 = 0.0f;
#pragma unroll
  for (int i = 0; i < 4; ++i) { float d = v[i] - mean; s2 += d * d; }
  red[tid] = s2; __syncthreads();
  for (int off = 64; off > 0; off >>= 1) { if (tid < off) red[tid] += red[tid + off]; __syncthreads(); }
  float inv = rsqrtf(red[0] * (1.0f / D_) + 1e-5f);
#pragma unroll
  for (int i = 0; i < 4; ++i) {
    int d = i * 128 + tid;
    out[(size_t)row * D_ + d] = (_Float16)((v[i] - mean) * inv * gs[d] + gb[d]);
  }
}

// ---------------- generic WMMA GEMM ----------------
// out[M,Nout] = A[M,K](f16) @ W[K,Nout] + bias, with W given transposed (WT [Nout,K] f16).
// Optional GELU, optional f32 residual add, optional f32 and/or f16 outputs.
// Block: 256 threads = 8 waves as 2(M)x4(N); block tile 64x256; wave tile 32x64.
__global__ void __launch_bounds__(256)
gemm_wmma_kernel(const _Float16* A, const _Float16* WT, const float* bias,
                 const float* res, float* outF, _Float16* outH,
                 int M, int K, int Nout, int doGelu) {
  int tid = threadIdx.x;
  int wave = tid >> 5, lane = tid & 31;
  int wm = wave & 1;            // 0..1
  int wn = wave >> 1;           // 0..3
  int m0 = blockIdx.x * 64 + wm * 32;
  int n0 = blockIdx.y * 256 + wn * 64;
  v8f acc[2][4] = {};
  for (int k0 = 0; k0 < K; k0 += 32) {
    if (k0 + 32 < K)
      __builtin_prefetch(A + (size_t)(m0 + (lane & 15)) * K + k0 + 32, 0, 1);
    v16h a0 = load_a_frag(A, K, m0, k0);
    v16h a1 = load_a_frag(A, K, m0 + 16, k0);
#pragma unroll
    for (int ni = 0; ni < 4; ++ni) {
      v16h b = load_b_frag(WT, K, n0 + ni * 16, k0);
      acc[0][ni] = wmma_f16(a0, b, acc[0][ni]);
      acc[1][ni] = wmma_f16(a1, b, acc[1][ni]);
    }
  }
  int prow = (lane & 16) ? 8 : 0;
  int pcol = lane & 15;
#pragma unroll
  for (int mi = 0; mi < 2; ++mi)
#pragma unroll
    for (int ni = 0; ni < 4; ++ni) {
      int col = n0 + ni * 16 + pcol;
      float bb = bias[col];
#pragma unroll
      for (int e = 0; e < 8; ++e) {
        int row = m0 + mi * 16 + e + prow;
        size_t idx = (size_t)row * Nout + col;
        float v = acc[mi][ni][e] + bb;
        if (doGelu) v = gelu_f(v);
        if (res)  v += res[idx];
        if (outF) outF[idx] = v;
        if (outH) outH[idx] = (_Float16)v;
      }
    }
}

// ---------------- flash attention (WMMA for QK^T and P*V) ----------------
// grid: (ceil(N/64), B*NH); block 128 = 4 waves, each wave owns 16 query rows.
__global__ void __launch_bounds__(128)
attn_kernel(const _Float16* qkv, _Float16* attO) {
  __shared__ __align__(16) _Float16 sK [64][80];      // K tile, row-major [j][d]
  __shared__ __align__(16) _Float16 sVt[64][80];      // V^T tile: [d][j]
  __shared__ __align__(16) _Float16 sP [4][16][80];   // per-wave P tile: [row][j]
  int bh = blockIdx.y;
  int b = bh >> 3, h = bh & 7;
  int q0 = blockIdx.x * 64;
  int tid = threadIdx.x;
  int wave = tid >> 5, lane = tid & 31;
  int i0 = q0 + wave * 16;
  const _Float16* qbase = qkv + (size_t)b * NTOK * (3 * D_) + h * DH_;
  const _Float16* kbase = qbase + D_;
  const _Float16* vbase = qbase + 2 * D_;

  v8f accO[4] = {};
  float mrun[8], lrun[8];
#pragma unroll
  for (int e = 0; e < 8; ++e) { mrun[e] = -1e30f; lrun[e] = 0.0f; }

  int prow = (lane & 16) ? 8 : 0;
  int pcol = lane & 15;
  const int nkt = (NTOK + 63) / 64;   // 29

  for (int kt = 0; kt < nkt; ++kt) {
    int k0 = kt * 64;
    __syncthreads();
    // ---- stage K tile row-major into sK (async DMA to LDS when available) ----
#pragma unroll
    for (int c = 0; c < 4; ++c) {
      int idx = c * 128 + tid;           // 0..511 -> 64 rows x 8 chunks of 8 halves
      int j = idx >> 3, cc = idx & 7;
      int row = k0 + j; if (row > NTOK - 1) row = NTOK - 1;
      const _Float16* src = kbase + (size_t)row * (3 * D_) + cc * 8;
#if HAVE_ASYNC_LDS
      async_copy16(src, &sK[j][cc * 8]);
#else
      *(v8h*)&sK[j][cc * 8] = *(const v8h*)src;
#endif
    }
    // ---- stage V tile transposed into sVt (vector global reads, scalar LDS writes) ----
#pragma unroll
    for (int c = 0; c < 4; ++c) {
      int idx = c * 128 + tid;
      int j = idx >> 3, cc = idx & 7;
      int row = k0 + j;
      bool ok = row < NTOK; if (!ok) row = NTOK - 1;
      v8h vv = *(const v8h*)(vbase + (size_t)row * (3 * D_) + cc * 8);
#pragma unroll
      for (int e = 0; e < 8; ++e) sVt[cc * 8 + e][j] = ok ? vv[e] : (_Float16)0.0f;
    }
#if HAVE_ASYNC_LDS
    __builtin_amdgcn_s_wait_asynccnt(0);
#endif
    __syncthreads();

    // S = (Q K^T) * scale, masked. B fragments come from the shared K tile.
    v8f accS[4] = {};
    for (int ks = 0; ks < DH_; ks += 32) {
      v16h afr = load_a_frag_clamp(qbase, 3 * D_, i0, ks, NTOK - 1);
#pragma unroll
      for (int jt = 0; jt < 4; ++jt) {
        v16h bfr = load_b_frag(&sK[0][0], 80, jt * 16, ks);
        accS[jt] = wmma_f16(afr, bfr, accS[jt]);
      }
    }
#pragma unroll
    for (int jt = 0; jt < 4; ++jt) {
      bool valid = (k0 + jt * 16 + pcol) < NTOK;
#pragma unroll
      for (int e = 0; e < 8; ++e) {
        float s = accS[jt][e] * 0.125f;        // 1/sqrt(64)
        accS[jt][e] = valid ? s : -1e30f;
      }
    }
    // online softmax: row reductions across the 16 lanes of each half-wave
    float mx[8];
#pragma unroll
    for (int e = 0; e < 8; ++e)
      mx[e] = fmaxf(fmaxf(accS[0][e], accS[1][e]), fmaxf(accS[2][e], accS[3][e]));
#pragma unroll
    for (int m = 1; m < 16; m <<= 1)
#pragma unroll
      for (int e = 0; e < 8; ++e) mx[e] = fmaxf(mx[e], __shfl_xor(mx[e], m, 32));
    float corr[8];
#pragma unroll
    for (int e = 0; e < 8; ++e) {
      float mn = fmaxf(mrun[e], mx[e]);
      corr[e] = __expf(mrun[e] - mn);
      mrun[e] = mn;
    }
    float rs[8];
#pragma unroll
    for (int e = 0; e < 8; ++e) rs[e] = 0.0f;
#pragma unroll
    for (int jt = 0; jt < 4; ++jt)
#pragma unroll
      for (int e = 0; e < 8; ++e) {
        float pv = __expf(accS[jt][e] - mrun[e]);
        accS[jt][e] = pv;
        rs[e] += pv;
      }
#pragma unroll
    for (int m = 1; m < 16; m <<= 1)
#pragma unroll
      for (int e = 0; e < 8; ++e) rs[e] += __shfl_xor(rs[e], m, 32);
#pragma unroll
    for (int e = 0; e < 8; ++e) lrun[e] = lrun[e] * corr[e] + rs[e];
#pragma unroll
    for (int dt = 0; dt < 4; ++dt)
#pragma unroll
      for (int e = 0; e < 8; ++e) accO[dt][e] *= corr[e];
    // spill P (C layout) to LDS so it can be re-read as an A fragment
#pragma unroll
    for (int jt = 0; jt < 4; ++jt)
#pragma unroll
      for (int e = 0; e < 8; ++e)
        sP[wave][prow + e][jt * 16 + pcol] = (_Float16)accS[jt][e];
    // O += P * V  (B fragments from transposed V in LDS)
    for (int ks = 0; ks < 64; ks += 32) {
      v16h pf = load_a_frag(&sP[wave][0][0], 80, 0, ks);
#pragma unroll
      for (int dt = 0; dt < 4; ++dt) {
        v16h vf = load_b_frag(&sVt[0][0], 80, dt * 16, ks);
        accO[dt] = wmma_f16(pf, vf, accO[dt]);
      }
    }
  }

  // epilogue: O /= l, store f16
  float inv[8];
#pragma unroll
  for (int e = 0; e < 8; ++e) inv[e] = 1.0f / lrun[e];
#pragma unroll
  for (int dt = 0; dt < 4; ++dt)
#pragma unroll
    for (int e = 0; e < 8; ++e) {
      int row = i0 + e + prow;
      if (row < NTOK) {
        int col = h * DH_ + dt * 16 + pcol;
        attO[((size_t)b * NTOK + row) * D_ + col] = (_Float16)(accO[dt][e] * inv[e]);
      }
    }
}

// ---------------- head + L1 loss ----------------
__global__ void __launch_bounds__(256)
head_loss_kernel(const float* x, const float* patches, const int* mIdx,
                 const float* Wt, const float* bt, float* out) {
  __shared__ float red[256];
  int t = blockIdx.x * 256 + threadIdx.x;
  float loss = 0.0f;
  if (t < B_ * NM_) {
    int b = t / NM_;
    int n = mIdx[t];
    const float* xr = x + ((size_t)b * NTOK + n) * D_;
    float s[PD_];
#pragma unroll
    for (int p = 0; p < PD_; ++p) s[p] = bt[p];
    for (int d = 0; d < D_; ++d) {
      float xv = xr[d];
#pragma unroll
      for (int p = 0; p < PD_; ++p) s[p] += xv * Wt[d * PD_ + p];
    }
    const float* mp = patches + ((size_t)b * NTOK + n) * PD_;
#pragma unroll
    for (int p = 0; p < PD_; ++p) loss += fabsf(s[p] - mp[p]);
  }
  red[threadIdx.x] = loss; __syncthreads();
  for (int off = 128; off > 0; off >>= 1) {
    if (threadIdx.x < off) red[threadIdx.x] += red[threadIdx.x + off];
    __syncthreads();
  }
  if (threadIdx.x == 0) atomicAdd(out, red[0] * LOSS_SCALE);
}

// ---------------- host launcher ----------------
extern "C" void kernel_launch(void* const* d_in, const int* in_sizes, int n_in,
                              void* d_out, int out_size, void* d_ws, size_t ws_size,
                              hipStream_t stream) {
  const float* img        = (const float*)d_in[0];
  const float* pos_table  = (const float*)d_in[1];
  const float* val_table  = (const float*)d_in[2];
  const float* patch_W    = (const float*)d_in[3];
  const float* patch_b    = (const float*)d_in[4];
  const float* mask_token = (const float*)d_in[5];
  const float* ln1_s      = (const float*)d_in[6];
  const float* ln1_b      = (const float*)d_in[7];
  const float* Wqkv       = (const float*)d_in[8];
  const float* bqkv       = (const float*)d_in[9];
  const float* Wo         = (const float*)d_in[10];
  const float* bo         = (const float*)d_in[11];
  const float* ln2_s      = (const float*)d_in[12];
  const float* ln2_b      = (const float*)d_in[13];
  const float* W1         = (const float*)d_in[14];
  const float* b1         = (const float*)d_in[15];
  const float* W2         = (const float*)d_in[16];
  const float* b2         = (const float*)d_in[17];
  const float* Wt         = (const float*)d_in[18];
  const float* bt         = (const float*)d_in[19];
  const int*   vlen       = (const int*)d_in[20];
  const int*   mIdx       = (const int*)d_in[21];
  float* out = (float*)d_out;

  size_t off = 0;
  char* wsb = (char*)d_ws;
  auto take = [&](size_t bytes) -> void* {
    void* p = wsb + off;
    off += (bytes + 255) & ~(size_t)255;
    return p;
  };
  float*     x     = (float*)    take((size_t)BN_ * D_ * 4);
  float*     pat   = (float*)    take((size_t)BN_ * PD_ * 4);
  int*       msk   = (int*)      take((size_t)BN_ * 4);
  _Float16*  h16   = (_Float16*) take((size_t)BN_ * D_ * 2);
  _Float16*  qkv16 = (_Float16*) take((size_t)BN_ * 3 * D_ * 2);
  _Float16*  att16 = (_Float16*) take((size_t)BN_ * D_ * 2);
  _Float16*  mlp16 = (_Float16*) take((size_t)BN_ * MLP_ * 2);
  _Float16*  WqkvT = (_Float16*) take((size_t)L_ * 3 * D_ * D_ * 2);
  _Float16*  WoT   = (_Float16*) take((size_t)L_ * D_ * D_ * 2);
  _Float16*  W1T   = (_Float16*) take((size_t)L_ * MLP_ * D_ * 2);
  _Float16*  W2T   = (_Float16*) take((size_t)L_ * D_ * MLP_ * 2);
  (void)n_in; (void)in_sizes; (void)out_size; (void)ws_size;

  // setup
  init_kernel<<<(BN_ + 255) / 256, 256, 0, stream>>>(msk, out, BN_);
  scatter_mask_kernel<<<(B_ * NM_ + 255) / 256, 256, 0, stream>>>(msk, mIdx);
  wtrans_kernel<<<1024, 256, 0, stream>>>(Wqkv, WqkvT, D_, 3 * D_, L_ * D_ * 3 * D_);
  wtrans_kernel<<<1024, 256, 0, stream>>>(Wo,   WoT,   D_, D_,     L_ * D_ * D_);
  wtrans_kernel<<<1024, 256, 0, stream>>>(W1,   W1T,   D_, MLP_,   L_ * D_ * MLP_);
  wtrans_kernel<<<1024, 256, 0, stream>>>(W2,   W2T,   MLP_, D_,   L_ * MLP_ * D_);

  embed_kernel<<<BN_, 128, 0, stream>>>(img, pos_table, val_table, patch_W, patch_b,
                                        mask_token, vlen, msk, x, pat);

  for (int l = 0; l < L_; ++l) {
    // LN1 -> h16
    ln_kernel<<<BN_, 128, 0, stream>>>(x, ln1_s + l * D_, ln1_b + l * D_, h16);
    // qkv = h @ Wqkv + bqkv  (f16 out)
    gemm_wmma_kernel<<<dim3(BN_ / 64, (3 * D_) / 256), 256, 0, stream>>>(
        h16, WqkvT + (size_t)l * 3 * D_ * D_, bqkv + l * 3 * D_,
        nullptr, nullptr, qkv16, BN_, D_, 3 * D_, 0);
    // attention -> att16
    attn_kernel<<<dim3((NTOK + 63) / 64, B_ * NH_), 128, 0, stream>>>(qkv16, att16);
    // x += att @ Wo + bo
    gemm_wmma_kernel<<<dim3(BN_ / 64, D_ / 256), 256, 0, stream>>>(
        att16, WoT + (size_t)l * D_ * D_, bo + l * D_,
        x, x, nullptr, BN_, D_, D_, 0);
    // LN2 -> h16
    ln_kernel<<<BN_, 128, 0, stream>>>(x, ln2_s + l * D_, ln2_b + l * D_, h16);
    // mlp16 = gelu(h @ W1 + b1)  (f16 out)
    gemm_wmma_kernel<<<dim3(BN_ / 64, MLP_ / 256), 256, 0, stream>>>(
        h16, W1T + (size_t)l * MLP_ * D_, b1 + l * MLP_,
        nullptr, nullptr, mlp16, BN_, D_, MLP_, 1);
    // x += mlp16 @ W2 + b2
    gemm_wmma_kernel<<<dim3(BN_ / 64, D_ / 256), 256, 0, stream>>>(
        mlp16, W2T + (size_t)l * D_ * MLP_, b2 + l * D_,
        x, x, nullptr, BN_, MLP_, D_, 0);
  }

  head_loss_kernel<<<(B_ * NM_ + 255) / 256, 256, 0, stream>>>(x, pat, mIdx, Wt, bt, out);
}
